// ShapeNetPnet2Encoder_44839458570447
// MI455X (gfx1250) — matrix operand order repaired and checked by reading, not compiled, not run on hardware
//
#include <hip/hip_runtime.h>

typedef __attribute__((ext_vector_type(16))) _Float16 v16h;
typedef __attribute__((ext_vector_type(8)))  _Float16 v8h;
typedef __attribute__((ext_vector_type(8)))  float    v8f;

#define BN_INV 0.99999499f  /* 1/sqrt(1+1e-5) */

// ---------------------------------------------------------------------------
// (B,3,N) -> (B,N,3)
__global__ void k_transpose(const float* __restrict__ x, float* __restrict__ y,
                            int Nc, long total) {
  long i = (long)blockIdx.x * blockDim.x + threadIdx.x;   // over B*N
  if (i >= total) return;
  long b = i / Nc, n = i % Nc;
  y[i * 3 + 0] = x[(b * 3 + 0) * (long)Nc + n];
  y[i * 3 + 1] = x[(b * 3 + 1) * (long)Nc + n];
  y[i * 3 + 2] = x[(b * 3 + 2) * (long)Nc + n];
}

// ---------------------------------------------------------------------------
// Farthest point sampling: one block (256 threads) per batch. dist kept in
// registers (<=32 points/thread), argmax via LDS tree (ties -> lowest index,
// matching jnp.argmax).
#define FPS_TPB 256
__global__ void k_fps(const float* __restrict__ pts, int N, int npoint,
                      int* __restrict__ fidx) {
  int b = blockIdx.x;
  int t = threadIdx.x;
  int per = (N + FPS_TPB - 1) / FPS_TPB;
  if (per > 32) per = 32;
  float dist[32];
  #pragma unroll
  for (int j = 0; j < 32; ++j) dist[j] = 1e10f;

  __shared__ float s_val[FPS_TPB];
  __shared__ int   s_idx[FPS_TPB];
  __shared__ float s_c[3];
  int far = 0;
  for (int it = 0; it < npoint; ++it) {
    if (t == 0) {
      fidx[b * npoint + it] = far;
      s_c[0] = pts[((long)b * N + far) * 3 + 0];
      s_c[1] = pts[((long)b * N + far) * 3 + 1];
      s_c[2] = pts[((long)b * N + far) * 3 + 2];
    }
    __syncthreads();
    float cx = s_c[0], cy = s_c[1], cz = s_c[2];
    float best = -1.0f; int bi = 0x7fffffff;
    for (int j = 0; j < per; ++j) {
      int i = j * FPS_TPB + t;
      if (i < N) {
        float dx = pts[((long)b * N + i) * 3 + 0] - cx;
        float dy = pts[((long)b * N + i) * 3 + 1] - cy;
        float dz = pts[((long)b * N + i) * 3 + 2] - cz;
        float d = dx * dx + dy * dy + dz * dz;
        float nd = fminf(dist[j], d);
        dist[j] = nd;
        if (nd > best || (nd == best && i < bi)) { best = nd; bi = i; }
      }
    }
    s_val[t] = best; s_idx[t] = bi;
    __syncthreads();
    for (int off = FPS_TPB / 2; off > 0; off >>= 1) {
      if (t < off) {
        float ov = s_val[t + off]; int oi = s_idx[t + off];
        if (ov > s_val[t] || (ov == s_val[t] && oi < s_idx[t])) {
          s_val[t] = ov; s_idx[t] = oi;
        }
      }
      __syncthreads();
    }
    far = s_idx[0];
  }
}

// ---------------------------------------------------------------------------
// new_xyz = xyz[fidx]
__global__ void k_gather3(const float* __restrict__ xyz, const int* __restrict__ idx,
                          float* __restrict__ out, int N, int S, int Bc) {
  int i = blockIdx.x * blockDim.x + threadIdx.x;   // over B*S
  if (i >= Bc * S) return;
  int b = i / S;
  int id = idx[i];
  out[i * 3 + 0] = xyz[((long)b * N + id) * 3 + 0];
  out[i * 3 + 1] = xyz[((long)b * N + id) * 3 + 1];
  out[i * 3 + 2] = xyz[((long)b * N + id) * 3 + 2];
}

// ---------------------------------------------------------------------------
// Ball-query + group: one wave32 per (b,s). Scan points in index order,
// ballot in-radius mask, prefix-popcount assigns the first 32 hits to lanes
// (== reference sort-then-take semantics); pad with first hit. Each lane
// writes one grouped row: [xyz - center | feat].
__global__ void k_group(const float* __restrict__ xyz, const float* __restrict__ feats,
                        const float* __restrict__ new_xyz, float* __restrict__ out,
                        int N, int S, int Cf, float r2, int Bc) {
  int gwave = (int)((blockIdx.x * (long)blockDim.x + threadIdx.x) >> 5);
  int lane = threadIdx.x & 31;
  if (gwave >= Bc * S) return;
  int b = gwave / S, s = gwave % S;
  float cx = new_xyz[(b * S + s) * 3 + 0];
  float cy = new_xyz[(b * S + s) * 3 + 1];
  float cz = new_xyz[(b * S + s) * 3 + 2];

  int sel = -1, count = 0;
  for (int base = 0; base < N && count < 32; base += 32) {
    int i = base + lane;
    bool inr = false;
    if (i < N) {
      float dx = xyz[((long)b * N + i) * 3 + 0] - cx;
      float dy = xyz[((long)b * N + i) * 3 + 1] - cy;
      float dz = xyz[((long)b * N + i) * 3 + 2] - cz;
      inr = (dx * dx + dy * dy + dz * dz <= r2);
    }
    unsigned mask = (unsigned)__ballot(inr);
    int cnt = __popc(mask);
    int tgt = lane - count;
    if (sel < 0 && tgt >= 0 && tgt < cnt) {
      int pos = 0;
      for (int j = 0; j < 32; ++j) {
        if ((mask >> j) & 1u) { if (tgt == 0) { pos = j; break; } --tgt; }
      }
      sel = base + pos;
    }
    count += cnt;
  }
  int first = __shfl(sel, 0, 32);
  if (sel < 0) sel = (first < 0) ? 0 : first;

  long row = ((long)(b * S + s)) * 32 + lane;
  int C0 = 3 + Cf;
  out[row * C0 + 0] = xyz[((long)b * N + sel) * 3 + 0] - cx;
  out[row * C0 + 1] = xyz[((long)b * N + sel) * 3 + 1] - cy;
  out[row * C0 + 2] = xyz[((long)b * N + sel) * 3 + 2] - cz;
  for (int c = 0; c < Cf; ++c)
    out[row * C0 + 3 + c] = feats[((long)b * N + sel) * Cf + c];
}

// ---------------------------------------------------------------------------
// f32 [M,K] -> f16 [Mp,Kp] zero-padded (Kp mult of 32, Mp mult of block tile).
// Pushes all bounds handling out of the GEMM hot loop.
__global__ void k_cvt_pad(const float* __restrict__ in, _Float16* __restrict__ out,
                          long M, int K, long Mp, int Kp) {
  long i = (long)blockIdx.x * blockDim.x + threadIdx.x;
  if (i >= Mp * (long)Kp) return;
  long m = i / Kp; int k = (int)(i % Kp);
  out[i] = (m < M && k < K) ? (_Float16)in[m * (long)K + k] : (_Float16)0.0f;
}

// ---------------------------------------------------------------------------
// WMMA GEMM on pre-padded f16 operands: C[M,N] = epilogue(A[Mp,Kp] * W[Np,Kp]^T).
// 256 threads = 8 waves. Block tile 128(M) x 64(N): each wave owns a 16-row
// strip and computes FOUR 16x16 N-tiles, reusing its A fragment across 4
// v_wmma_f32_16x16x32_f16 per K-step. Staging is branch-free 128-bit copies
// (global_load_b128 -> ds_store_b128); bounds checks only on the f32 store.
#define GT_M 128
#define GT_N 64
__global__ void __launch_bounds__(256)
k_gemm_wmma(const _Float16* __restrict__ A, const _Float16* __restrict__ W,
            const float* __restrict__ bias, const float* __restrict__ gamma,
            const float* __restrict__ beta, float bninv, int relu,
            float* __restrict__ Cout, int M, int Nn, int Kp) {
  __shared__ v8h sA8[GT_M * 4];   // 128 rows x 32 f16 = 8 KB
  __shared__ v8h sB8[GT_N * 4];   // 64 rows x 32 f16 = 4 KB
  const _Float16* sA = (const _Float16*)sA8;
  const _Float16* sB = (const _Float16*)sB8;
  int tid = threadIdx.x;
  int wave = tid >> 5, lane = tid & 31;
  int mWave = blockIdx.x * GT_M + wave * 16;
  int nBase = blockIdx.y * GT_N;
  v8f acc[4] = {};

  const _Float16* Abase = A + (long)blockIdx.x * GT_M * Kp;
  const _Float16* Wbase = W + (long)nBase * Kp;

  for (int k0 = 0; k0 < Kp; k0 += 32) {
    // stage A tile: 512 x v8h chunks, 2 per thread, no guards (pre-padded)
    #pragma unroll
    for (int c = tid; c < GT_M * 4; c += 256) {
      int m = c >> 2, kc = c & 3;
      sA8[c] = *(const v8h*)(Abase + (long)m * Kp + k0 + kc * 8);
    }
    // stage B tile: 256 x v8h chunks, 1 per thread
    {
      int c = tid;
      int n = c >> 2, kc = c & 3;
      sB8[c] = *(const v8h*)(Wbase + (long)n * Kp + k0 + kc * 8);
    }
    // prefetch next K-step of our A strip (global_prefetch_b8)
    if (k0 + 32 < Kp)
      __builtin_prefetch(Abase + (long)(wave * 16 + (lane & 15)) * Kp + k0 + 32, 0, 1);
    __syncthreads();

    int hf = lane >> 4;
    int mloc = wave * 16 + (lane & 15);
    v16h a;
    #pragma unroll
    for (int j = 0; j < 16; ++j) {           // A: 16-bit 16x32 striping (ISA 7.12.2)
      int v = j >> 1, sub = j & 1;
      int k = ((v < 4) ? (v * 2) : ((v - 4) * 2 + 16)) + hf * 8 + sub;
      a[j] = sA[mloc * 32 + k];
    }
    #pragma unroll
    for (int t = 0; t < 4; ++t) {
      int nloc = t * 16 + (lane & 15);
      v16h bf;
      #pragma unroll
      for (int j = 0; j < 16; ++j) {         // B: same 16-bit K striping as A
        int k = ((j < 8) ? j : (j + 8)) + hf * 8;
        bf[j] = sB[nloc * 32 + k];
      }
      acc[t] = __builtin_amdgcn_wmma_f32_16x16x32_f16(
          false, a, false, bf, (short)0, acc[t], false, false);
    }
    __syncthreads();
  }

  int mBase = mWave + (lane >> 4) * 8;
  #pragma unroll
  for (int t = 0; t < 4; ++t) {
    int n = nBase + t * 16 + (lane & 15);
    if (n < Nn) {
      float bb = bias  ? bias[n]          : 0.0f;
      float gg = gamma ? gamma[n] * bninv : 1.0f;
      float be = beta  ? beta[n]          : 0.0f;
      #pragma unroll
      for (int r = 0; r < 8; ++r) {
        int m = mBase + r;
        if (m < M) {
          float x = (acc[t][r] + bb) * gg + be;
          if (relu) x = fmaxf(x, 0.0f);
          Cout[(long)m * Nn + n] = x;
        }
      }
    }
  }
}

// ---------------------------------------------------------------------------
// max over groups of g consecutive rows: in[rowsOut*g, C] -> out[rowsOut, C]
__global__ void k_maxpool(const float* __restrict__ in, float* __restrict__ out,
                          long rowsOut, int C, int g) {
  long i = (long)blockIdx.x * blockDim.x + threadIdx.x;
  if (i >= rowsOut * (long)C) return;
  long r = i / C; int c = (int)(i % C);
  const float* p = in + r * (long)g * C + c;
  float m = p[0];
  for (int j = 1; j < g; ++j) m = fmaxf(m, p[(long)j * C]);
  out[i] = m;
}

// ---------------------------------------------------------------------------
// group-all concat: out[rows, 3+Cf] = [xyz | f]
__global__ void k_concat(const float* __restrict__ xyz, const float* __restrict__ f,
                         float* __restrict__ out, int rows, int Cf) {
  int C0 = 3 + Cf;
  long i = (long)blockIdx.x * blockDim.x + threadIdx.x;
  if (i >= (long)rows * C0) return;
  int r = (int)(i / C0), c = (int)(i % C0);
  out[i] = (c < 3) ? xyz[r * 3 + c] : f[(long)r * Cf + (c - 3)];
}

// ---------------------------------------------------------------------------
static void run_gemm_layer(const float* inF32, long Mrows, int K, int Nn,
                           const float* W, const float* bias,
                           const float* gamma, const float* beta, int relu,
                           float* outF32, _Float16* aH, _Float16* wH,
                           hipStream_t stream) {
  int  Kp = (K + 31) & ~31;
  long Mp = ((Mrows + GT_M - 1) / GT_M) * (long)GT_M;
  long Np = ((Nn + GT_N - 1) / GT_N) * (long)GT_N;
  long na = Mp * Kp;
  k_cvt_pad<<<(unsigned)((na + 255) / 256), 256, 0, stream>>>(inF32, aH, Mrows, K, Mp, Kp);
  long nw = Np * Kp;
  k_cvt_pad<<<(unsigned)((nw + 255) / 256), 256, 0, stream>>>(W, wH, Nn, K, Np, Kp);
  dim3 grid((unsigned)(Mp / GT_M), (unsigned)(Np / GT_N));
  k_gemm_wmma<<<grid, 256, 0, stream>>>(aH, wH, bias, gamma, beta, BN_INV, relu,
                                        outF32, (int)Mrows, Nn, Kp);
}

extern "C" void kernel_launch(void* const* d_in, const int* in_sizes, int n_in,
                              void* d_out, int out_size, void* d_ws, size_t ws_size,
                              hipStream_t stream) {
  (void)in_sizes; (void)n_in; (void)out_size; (void)ws_size;
  const int Bc = 8, N0 = 8192, NS = 32;
  const float* xyz_in = (const float*)d_in[0];

  char* wsb = (char*)d_ws;
  size_t off = 0;
  auto alloc = [&](size_t bytes) -> void* {
    void* p = (void*)(wsb + off);
    off += (bytes + 255) & ~(size_t)255;
    return p;
  };

  float*    pts  = (float*)alloc((size_t)Bc * N0 * 3 * 4);
  float*    big0 = (float*)alloc((size_t)17 * 1024 * 1024 * 4);   // 17M f32
  float*    big1 = (float*)alloc((size_t)17 * 1024 * 1024 * 4);
  _Float16* aH   = (_Float16*)alloc((size_t)16 * 1024 * 1024 * 2);
  _Float16* wH   = (_Float16*)alloc((size_t)40 * 1024 * 1024 * 2);

  {
    long tot = (long)Bc * N0;
    k_transpose<<<(unsigned)((tot + 255) / 256), 256, 0, stream>>>(xyz_in, pts, N0, tot);
  }

  const int   S_arr[4]  = {1024, 512, 128, 16};
  const float r2_arr[4] = {0.01f, 0.04f, 0.16f, 0.64f};
  const int   mlp[5][3] = {{32,32,64},{64,64,128},{128,128,256},{256,256,512},{512,512,1024}};

  // d_in layout: [0]=xyz; then sa1..sa5: 3 sublayers x (W,b,gamma,beta); head: 4 x (W,b)
  int p = 1;

  const float* cur_xyz  = pts;
  const float* cur_feat = pts;
  int curN = N0, curCf = 3;

  for (int li = 0; li < 4; ++li) {
    int S = S_arr[li];
    int* fidx = (int*)alloc((size_t)Bc * S * 4);
    k_fps<<<Bc, FPS_TPB, 0, stream>>>(cur_xyz, curN, S, fidx);

    float* new_xyz = (float*)alloc((size_t)Bc * S * 3 * 4);
    k_gather3<<<(Bc * S + 255) / 256, 256, 0, stream>>>(cur_xyz, fidx, new_xyz, curN, S, Bc);

    int C0 = 3 + curCf;
    long rows = (long)Bc * S * NS;
    float* grouped = big0;
    {
      int waves = Bc * S;
      k_group<<<(waves * 32 + 255) / 256, 256, 0, stream>>>(
          cur_xyz, cur_feat, new_xyz, grouped, curN, S, curCf, r2_arr[li], Bc);
    }

    const float* curIn = grouped;
    float* bufs[2] = {big1, big0};
    int cin = C0, cout = 0;
    for (int j = 0; j < 3; ++j) {
      const float* W  = (const float*)d_in[p++];
      const float* bb = (const float*)d_in[p++];
      const float* gg = (const float*)d_in[p++];
      const float* be = (const float*)d_in[p++];
      cout = mlp[li][j];
      float* outb = bufs[j & 1];
      run_gemm_layer(curIn, rows, cin, cout, W, bb, gg, be, 1, outb, aH, wH, stream);
      curIn = outb; cin = cout;
    }

    float* pooled = (float*)alloc((size_t)Bc * S * cout * 4);
    k_maxpool<<<(unsigned)(((long)Bc * S * cout + 255) / 256), 256, 0, stream>>>(
        curIn, pooled, (long)Bc * S, cout, NS);

    cur_xyz = new_xyz; cur_feat = pooled; curN = S; curCf = cout;
  }

  // ---- SA5: group all (rows = B*16, C = 3+512) ----
  {
    int rows5 = Bc * curN;                 // 8*16 = 128
    k_concat<<<(unsigned)(((long)rows5 * (3 + curCf) + 255) / 256), 256, 0, stream>>>(
        cur_xyz, cur_feat, big0, rows5, curCf);

    const float* curIn = big0;
    float* bufs[2] = {big1, big0};
    int cin = 3 + curCf, cout = 0;
    for (int j = 0; j < 3; ++j) {
      const float* W  = (const float*)d_in[p++];
      const float* bb = (const float*)d_in[p++];
      const float* gg = (const float*)d_in[p++];
      const float* be = (const float*)d_in[p++];
      cout = mlp[4][j];
      float* outb = bufs[j & 1];
      run_gemm_layer(curIn, rows5, cin, cout, W, bb, gg, be, 1, outb, aH, wH, stream);
      curIn = outb; cin = cout;
    }

    float* h = (float*)alloc((size_t)Bc * cout * 4);
    k_maxpool<<<(unsigned)(((long)Bc * cout + 255) / 256), 256, 0, stream>>>(
        curIn, h, Bc, cout, curN);
    cur_feat = h; curCf = cout;            // (8, 1024)
  }

  // ---- head: 1024->1024 (relu), 1024->1024 (relu), 1024->6144 (relu), 6144->6144 ----
  {
    const int hco[4] = {1024, 1024, 6144, 6144};
    const float* hin = cur_feat;
    float* bufs[2] = {big1, big0};
    int cin = curCf;
    for (int i = 0; i < 4; ++i) {
      const float* W  = (const float*)d_in[p++];
      const float* bb = (const float*)d_in[p++];
      int cout = hco[i];
      float* outb = (i == 3) ? (float*)d_out : bufs[i & 1];
      run_gemm_layer(hin, Bc, cin, cout, W, bb, nullptr, nullptr, (i < 3) ? 1 : 0,
                     outb, aH, wH, stream);
      hin = outb; cin = cout;
    }
  }
}
